// M2FMoEBlock_50680614093049
// MI455X (gfx1250) — matrix operand order repaired and checked by reading, not compiled, not run on hardware
//
#include <hip/hip_runtime.h>
#include <math.h>

// ---- problem constants (from reference) ----
#define B_    2
#define S_    1024
#define D_    512
#define E_    8
#define FF_   2048
#define NTOK  (B_ * S_)                 // 2048 tokens
#define CAPR  (NTOK * 2 + E_ * 128)     // 5120: gathered rows, worst-case padded

typedef __attribute__((ext_vector_type(16))) __bf16 v16bf;
typedef __attribute__((ext_vector_type(8)))  float  v8f;
typedef __attribute__((ext_vector_type(4)))  unsigned int v4u;
typedef __attribute__((ext_vector_type(8)))  int  v8i;
typedef __attribute__((ext_vector_type(4)))  int  v4i;

struct __align__(16) U4 { unsigned int x, y, z, w; };

#define LDST 40   // 32 bf16 data + 8 pad -> 80B row stride, 16B aligned

// ---------------------------------------------------------------------------
// helpers
// ---------------------------------------------------------------------------
__device__ __forceinline__ float gelu_tanh(float x) {
    const float c = 0.7978845608028654f;
    float t = tanhf(c * (x + 0.044715f * x * x * x));
    return 0.5f * x * (1.0f + t);
}

__device__ __forceinline__ float wave_sum(float v) {
#pragma unroll
    for (int o = 16; o > 0; o >>= 1) v += __shfl_xor(v, o, 32);
    return v;
}

// CDNA5 async copy engine: global -> LDS, 16B per lane, tracked by ASYNCcnt.
__device__ __forceinline__ void async_b128(unsigned lds_off, const void* gaddr) {
    asm volatile("global_load_async_to_lds_b128 %0, %1, off"
                 :: "v"(lds_off), "v"((unsigned long long)(size_t)gaddr)
                 : "memory");
}
__device__ __forceinline__ void wait_async2() {
    asm volatile("s_wait_asynccnt 0x2" ::: "memory");
}
__device__ __forceinline__ void wait_async0() {
    asm volatile("s_wait_asynccnt 0x0" ::: "memory");
}

// TDM: load a 128x32-element bf16 tile (row stride ldb elements) into LDS,
// inserting 16B padding after every 64B row => LDS row stride 80B.
// Descriptor layout per CDNA5 ISA sec. 8.3 (group0) / 8.4 (group1).
__device__ __forceinline__ void tdm_load_tile(const __bf16* gsrc,
                                              unsigned ldsOffBytes, int ldb)
{
    unsigned long long ga = (unsigned long long)(size_t)gsrc;
    v4u g0;
    g0[0] = 1u;                                   // count=1 (valid user D#)
    g0[1] = ldsOffBytes;                          // lds_addr
    g0[2] = (unsigned)(ga & 0xFFFFFFFFu);         // global_addr[31:0]
    g0[3] = (unsigned)((ga >> 32) & 0x01FFFFFFu)  // global_addr[56:32]
          | (2u << 30);                           // type=2 ("image")
    unsigned w0 = (1u << 16)                      // data_size=1 (2 bytes)
                | (1u << 20)                      // pad_enable
                | (3u << 22)                      // pad_interval: 16 DWORDs
                | (3u << 25);                     // pad_amount: 4 DWORDs
    unsigned td0 = 32u, td1 = 128u;               // tensor dims (no OOB)
    unsigned long long st0 = (unsigned long long)ldb;
    v8i g1;
    g1[0] = (int)w0;
    g1[1] = (int)((td0 & 0xFFFFu) << 16);                           // td0 lo
    g1[2] = (int)(((td0 >> 16) & 0xFFFFu) | ((td1 & 0xFFFFu) << 16));
    g1[3] = (int)(((td1 >> 16) & 0xFFFFu) | (32u << 16));           // tile_dim0=32
    g1[4] = (int)128u;                                              // tile_dim1=128
    g1[5] = (int)(st0 & 0xFFFFFFFFu);                               // dim0 stride
    g1[6] = (int)((st0 >> 32) & 0xFFFFu);
    g1[7] = 0;
    v4i g2 = {0, 0, 0, 0};
    v4i g3 = {0, 0, 0, 0};
    v8i g4 = {0, 0, 0, 0, 0, 0, 0, 0};   // 6-arg toolchain variant: zero-filled
    __builtin_amdgcn_tensor_load_to_lds(g0, g1, g2, g3, g4, 0);
}

// Stage one 128x32 bf16 A tile via per-lane async copies (2 b128 per thread).
__device__ __forceinline__ void stage_A_async(
    const __bf16* __restrict__ A, int lda, int r0, int k0, int M,
    unsigned aOffBytes, int tid)
{
#pragma unroll
    for (int i = 0; i < 2; ++i) {
        int c   = tid + 256 * i;          // 512 16B chunks per tile
        int row = c >> 2;
        int sub = c & 3;
        int gm  = r0 + row; gm = (gm < M) ? gm : 0;  // clamp (stores guarded)
        async_b128(aOffBytes + (unsigned)(row * 80 + sub * 16),
                   A + (size_t)gm * lda + k0 + sub * 8);
    }
}

// ---------------------------------------------------------------------------
// 128x128x32 bf16-WMMA GEMM core.  Double-buffered LDS pipeline:
//   A tiles: per-lane async global->LDS copies (ASYNCcnt)
//   B tiles: Tensor Data Mover 2D tile loads, issued by wave 0 (TENSORcnt)
// 8 waves; wave w owns rows [16w,16w+16), 8 accumulator tiles along N.
// ---------------------------------------------------------------------------
__device__ __forceinline__ void gemm_core(
    const __bf16* __restrict__ A, int lda,
    const __bf16* __restrict__ Bt, int ldb,
    const float* __restrict__ bias,
    float* __restrict__ Cf, __bf16* __restrict__ Cb, int ldc,
    int M, int K, int act, int mtile, int ntile, __bf16* ls)
{
    const int tid  = threadIdx.x;
    const int lane = tid & 31;
    const int w    = tid >> 5;
    const int r0   = mtile * 128;
    const int n0   = ntile * 128;

    // integer element offsets into the single shared array (keeps AS3 / ds ops)
    const int ATILE = 128 * LDST;
    const unsigned lsBytes = (unsigned)(size_t)ls;   // LDS base byte offset

    v8f acc[8];
#pragma unroll
    for (int i = 0; i < 8; ++i) acc[i] = (v8f){0.f,0.f,0.f,0.f,0.f,0.f,0.f,0.f};

    const int m = lane & 15;   // A-frag row / B-frag col within 16x16 tile
    const int h = lane >> 4;   // half-wave K interleave
    const int nk = K >> 5;

    // prologue: tile 0 in flight on both engines
    stage_A_async(A, lda, r0, 0, M, lsBytes, tid);
    if (w == 0)
        tdm_load_tile(Bt + (size_t)n0 * ldb, lsBytes + 2u * ATILE * 2u, ldb);

    for (int i = 0; i < nk; ++i) {
        int cur = i & 1;
        int nxt = cur ^ 1;
        if (i + 1 < nk) {
            // issue tile i+1 into the other buffer (freed by the trailing
            // barrier of iteration i-1), then drain only tile i:
            // both engines complete in order per wave.
            stage_A_async(A, lda, r0, (i + 1) * 32, M,
                          lsBytes + (unsigned)(nxt * ATILE) * 2u, tid);
            if (w == 0)
                tdm_load_tile(Bt + (size_t)n0 * ldb + (i + 1) * 32,
                              lsBytes + (unsigned)((2 + nxt) * ATILE) * 2u, ldb);
            wait_async2();
            if (w == 0) __builtin_amdgcn_s_wait_tensorcnt(1);
        } else {
            wait_async0();
            if (w == 0) __builtin_amdgcn_s_wait_tensorcnt(0);
        }
        __syncthreads();                       // tile i visible to all waves

        const int aSel = cur * ATILE;
        const int bSel = (2 + cur) * ATILE;

        // A fragment: 16x32 bf16 (ISA 7.12.2 layout) -> 2x ds_load_b128
        v16bf af;
        U4* aq = (U4*)&af;
        int abase = aSel + (w * 16 + m) * LDST + 8 * h;
        aq[0] = *(const U4*)&ls[abase];
        aq[1] = *(const U4*)&ls[abase + 16];

        // B fragments processed in pairs with two live register sets so the
        // scheduler can overlap the second pair's ds loads with the first WMMA
        // (partial s_wait_dscnt instead of full drains).
#pragma unroll
        for (int nt = 0; nt < 8; nt += 2) {
            v16bf bf0, bf1;
            U4* b0 = (U4*)&bf0;
            U4* b1 = (U4*)&bf1;
            int bb0 = bSel + ((nt    ) * 16 + m) * LDST + 8 * h;
            int bb1 = bSel + ((nt + 1) * 16 + m) * LDST + 8 * h;
            b0[0] = *(const U4*)&ls[bb0];
            b0[1] = *(const U4*)&ls[bb0 + 16];
            b1[0] = *(const U4*)&ls[bb1];
            b1[1] = *(const U4*)&ls[bb1 + 16];
            acc[nt] = __builtin_amdgcn_wmma_f32_16x16x32_bf16(
                false, af, false, bf0, (short)0, acc[nt], false, false);
            acc[nt + 1] = __builtin_amdgcn_wmma_f32_16x16x32_bf16(
                false, af, false, bf1, (short)0, acc[nt + 1], false, false);
        }
        __syncthreads();                       // buffer reusable in iter i+1
    }

    // epilogue: C/D layout VGPR r -> M = r + 8*half, N = lane&15
#pragma unroll
    for (int nt = 0; nt < 8; ++nt) {
        int gn = n0 + nt * 16 + m;
        float bv = bias ? bias[gn] : 0.0f;
#pragma unroll
        for (int r = 0; r < 8; ++r) {
            int gm = r0 + w * 16 + r + 8 * h;
            if (gm < M) {
                float v = acc[nt][r] + bv;
                if (act) v = gelu_tanh(v);
                if (Cf) Cf[(size_t)gm * ldc + gn] = v;
                else    Cb[(size_t)gm * ldc + gn] = (__bf16)v;
            }
        }
    }
}

// ---------------------------------------------------------------------------
// kernels
// ---------------------------------------------------------------------------
__global__ __launch_bounds__(256) void k_gemm(
    const __bf16* A, int lda, const __bf16* Bt, int ldb, const float* bias,
    float* Cf, __bf16* Cb, int ldc, int M, int K, int act)
{
    __shared__ __align__(16) __bf16 ls[4 * 128 * LDST];
    gemm_core(A, lda, Bt, ldb, bias, Cf, Cb, ldc, M, K, act,
              blockIdx.y, blockIdx.x, ls);
}

__global__ __launch_bounds__(256) void k_moe_gemm(
    const __bf16* A, int lda, const __bf16* WtBase, size_t wstride, int ldb,
    const float* biasBase, int bstride, float* Cf, __bf16* Cb, int ldc,
    int K, int act, const int* __restrict__ off)
{
    __shared__ __align__(16) __bf16 ls[4 * 128 * LDST];
    int row0  = blockIdx.y * 128;
    int total = off[E_];
    if (row0 >= total) return;            // uniform early exit (padded segments)
    int e = 0;
    while (e < E_ - 1 && row0 >= off[e + 1]) ++e;
    gemm_core(A, lda, WtBase + wstride * (size_t)e, ldb,
              biasBase + (size_t)bstride * e, Cf, Cb, ldc, total, K, act,
              blockIdx.y, blockIdx.x, ls);
}

// f32 -> bf16
__global__ void k_cvt(const float* __restrict__ s, __bf16* __restrict__ d, int n)
{
    int i = blockIdx.x * blockDim.x + threadIdx.x;
    if (i < n) d[i] = (__bf16)s[i];
}

// batched f32 [K][N] -> bf16 transposed [N][K] (per-launch weight prep)
__global__ void k_cvt_tr(const float* __restrict__ s, __bf16* __restrict__ d,
                         int K, int N, int n)
{
    int i = blockIdx.x * blockDim.x + threadIdx.x;
    if (i >= n) return;
    int kn = K * N;
    int b = i / kn, r = i - b * kn;
    int k = r / N, nn = r - k * N;
    d[(size_t)b * kn + (size_t)nn * K + k] = (__bf16)s[i];
}

// avg-pool by 2 along sequence
__global__ void k_pool2(const float* __restrict__ src, float* __restrict__ dst,
                        int n, int sOut)
{
    int i = blockIdx.x * blockDim.x + threadIdx.x;
    if (i >= n) return;
    int d   = i % D_;
    int row = i / D_;
    int b   = row / sOut, so = row % sOut;
    size_t sr = ((size_t)b * 2 * sOut + 2 * so) * D_ + d;
    dst[i] = 0.5f * (src[sr] + src[sr + D_]);
}

// mr combine + residual + LN1 (wave-per-token), dual f32/bf16 output
__global__ __launch_bounds__(256) void k_mrln1(
    const float* __restrict__ x, const float* __restrict__ y0,
    const float* __restrict__ y1, const float* __restrict__ y2,
    const float* __restrict__ g, const float* __restrict__ bt,
    float* __restrict__ xo, __bf16* __restrict__ xob)
{
    int w = threadIdx.x >> 5, lane = threadIdx.x & 31;
    int t = blockIdx.x * 8 + w;
    int b = t >> 10, s = t & 1023;
    const float* px = x  + (size_t)t * D_;
    const float* p0 = y0 + (size_t)t * D_;
    const float* p1 = y1 + ((size_t)b * (S_ / 2) + (s >> 1)) * D_;
    const float* p2 = y2 + ((size_t)b * (S_ / 4) + (s >> 2)) * D_;
    float vals[16]; float sum = 0.f;
#pragma unroll
    for (int i = 0; i < 16; ++i) {
        int d = lane + 32 * i;
        float v = (p0[d] + p1[d] + p2[d]) * (1.0f / 3.0f) + px[d];
        vals[i] = v; sum += v;
    }
    float mean = wave_sum(sum) * (1.0f / D_);
    float sq = 0.f;
#pragma unroll
    for (int i = 0; i < 16; ++i) { float c = vals[i] - mean; sq += c * c; }
    float inv = rsqrtf(wave_sum(sq) * (1.0f / D_) + 1e-5f);
#pragma unroll
    for (int i = 0; i < 16; ++i) {
        int d = lane + 32 * i;
        float v = (vals[i] - mean) * inv * g[d] + bt[d];
        xo[(size_t)t * D_ + d]  = v;
        xob[(size_t)t * D_ + d] = (__bf16)v;
    }
}

// softmax(alpha)-weighted view combine + residual + LN2, dual output
__global__ __launch_bounds__(256) void k_viewln2(
    const float* __restrict__ x1, const float* __restrict__ vbuf,
    const float* __restrict__ alpha, const float* __restrict__ g,
    const float* __restrict__ bt, float* __restrict__ xo,
    __bf16* __restrict__ xob)
{
    int w = threadIdx.x >> 5, lane = threadIdx.x & 31;
    int t = blockIdx.x * 8 + w;
    float a0 = alpha[0], a1 = alpha[1], a2 = alpha[2];
    float mx = fmaxf(a0, fmaxf(a1, a2));
    float e0 = expf(a0 - mx), e1 = expf(a1 - mx), e2 = expf(a2 - mx);
    float inv3 = 1.0f / (e0 + e1 + e2);
    a0 = e0 * inv3; a1 = e1 * inv3; a2 = e2 * inv3;
    const float* v0 = vbuf + (size_t)t * D_;
    const float* v1 = vbuf + (size_t)NTOK * D_ + (size_t)t * D_;
    const float* v2 = vbuf + 2 * (size_t)NTOK * D_ + (size_t)t * D_;
    const float* px = x1 + (size_t)t * D_;
    float vals[16]; float sum = 0.f;
#pragma unroll
    for (int i = 0; i < 16; ++i) {
        int d = lane + 32 * i;
        float v = a0 * v0[d] + a1 * v1[d] + a2 * v2[d] + px[d];
        vals[i] = v; sum += v;
    }
    float mean = wave_sum(sum) * (1.0f / D_);
    float sq = 0.f;
#pragma unroll
    for (int i = 0; i < 16; ++i) { float c = vals[i] - mean; sq += c * c; }
    float inv = rsqrtf(wave_sum(sq) * (1.0f / D_) + 1e-5f);
#pragma unroll
    for (int i = 0; i < 16; ++i) {
        int d = lane + 32 * i;
        float v = (vals[i] - mean) * inv * g[d] + bt[d];
        xo[(size_t)t * D_ + d]  = v;
        xob[(size_t)t * D_ + d] = (__bf16)v;
    }
}

// gate logits + top-2 + router softmax + atomic slot assignment
__global__ __launch_bounds__(256) void k_route(
    const float* __restrict__ x2, const float* __restrict__ gw,
    const float* __restrict__ gb, int* __restrict__ cnt,
    int* __restrict__ texp, int* __restrict__ tslot, float* __restrict__ tw)
{
    int w = threadIdx.x >> 5, lane = threadIdx.x & 31;
    int t = blockIdx.x * 8 + w;
    float l[E_];
#pragma unroll
    for (int e = 0; e < E_; ++e) l[e] = 0.f;
    for (int d = lane; d < D_; d += 32) {
        float xv = x2[(size_t)t * D_ + d];
#pragma unroll
        for (int e = 0; e < E_; ++e) l[e] += xv * gw[d * E_ + e];
    }
#pragma unroll
    for (int e = 0; e < E_; ++e) l[e] = wave_sum(l[e]);
    if (lane == 0) {
#pragma unroll
        for (int e = 0; e < E_; ++e) l[e] += gb[e];
        int e0 = 0; float v0 = l[0];
#pragma unroll
        for (int e = 1; e < E_; ++e) if (l[e] > v0) { v0 = l[e]; e0 = e; }
        int e1 = (e0 == 0) ? 1 : 0; float v1 = l[e1];
#pragma unroll
        for (int e = 0; e < E_; ++e)
            if (e != e0 && l[e] > v1) { v1 = l[e]; e1 = e; }
        float p0 = 1.0f / (1.0f + expf(v1 - v0));
        float p1 = 1.0f - p0;
        int s0 = atomicAdd(&cnt[e0], 1);
        int s1 = atomicAdd(&cnt[e1], 1);
        texp[2 * t] = e0;     tslot[2 * t] = s0;     tw[2 * t] = p0;
        texp[2 * t + 1] = e1; tslot[2 * t + 1] = s1; tw[2 * t + 1] = p1;
    }
}

// per-expert segment offsets, padded to 128-row multiples
__global__ void k_offsets(const int* __restrict__ cnt, int* __restrict__ off)
{
    if (threadIdx.x == 0 && blockIdx.x == 0) {
        int run = 0;
        for (int e = 0; e < E_; ++e) {
            off[e] = run;
            run += ((cnt[e] + 127) / 128) * 128;
        }
        off[E_] = run;
    }
}

// gather bf16 token rows into padded per-expert segments
__global__ __launch_bounds__(128) void k_gather(
    const __bf16* __restrict__ x2b, const int* __restrict__ texp,
    const int* __restrict__ tslot, const int* __restrict__ off,
    __bf16* __restrict__ xg)
{
    int pair = blockIdx.x;
    int t = pair >> 1;
    int row = off[texp[pair]] + tslot[pair];
    const uint2* src = (const uint2*)(x2b + (size_t)t * D_);
    uint2* dst = (uint2*)(xg + (size_t)row * D_);
    dst[threadIdx.x] = src[threadIdx.x];   // 128 threads x 8B = 1KB row
}

// weighted expert-output combine + residual + LN3 -> final output
__global__ __launch_bounds__(256) void k_finln3(
    const float* __restrict__ x2, const float* __restrict__ ye,
    const int* __restrict__ texp, const int* __restrict__ tslot,
    const int* __restrict__ off, const float* __restrict__ tw,
    const float* __restrict__ g, const float* __restrict__ bt,
    float* __restrict__ out)
{
    int w = threadIdx.x >> 5, lane = threadIdx.x & 31;
    int t = blockIdx.x * 8 + w;
    int r0 = off[texp[2 * t]] + tslot[2 * t];
    int r1 = off[texp[2 * t + 1]] + tslot[2 * t + 1];
    float w0 = tw[2 * t], w1 = tw[2 * t + 1];
    const float* y0 = ye + (size_t)r0 * D_;
    const float* y1 = ye + (size_t)r1 * D_;
    const float* px = x2 + (size_t)t * D_;
    float vals[16]; float sum = 0.f;
#pragma unroll
    for (int i = 0; i < 16; ++i) {
        int d = lane + 32 * i;
        float v = w0 * y0[d] + w1 * y1[d] + px[d];
        vals[i] = v; sum += v;
    }
    float mean = wave_sum(sum) * (1.0f / D_);
    float sq = 0.f;
#pragma unroll
    for (int i = 0; i < 16; ++i) { float c = vals[i] - mean; sq += c * c; }
    float inv = rsqrtf(wave_sum(sq) * (1.0f / D_) + 1e-5f);
#pragma unroll
    for (int i = 0; i < 16; ++i) {
        int d = lane + 32 * i;
        out[(size_t)t * D_ + d] = (vals[i] - mean) * inv * g[d] + bt[d];
    }
}

__global__ void k_zero_f32(float* __restrict__ p, int n)
{
    int i = blockIdx.x * blockDim.x + threadIdx.x;
    if (i < n) p[i] = 0.f;
}

// ---------------------------------------------------------------------------
// launch
// ---------------------------------------------------------------------------
extern "C" void kernel_launch(void* const* d_in, const int* in_sizes, int n_in,
                              void* d_out, int out_size, void* d_ws, size_t ws_size,
                              hipStream_t stream)
{
    const float* x        = (const float*)d_in[0];
    const float* mr_w     = (const float*)d_in[1];
    const float* mr_b     = (const float*)d_in[2];
    const float* mv_w     = (const float*)d_in[3];
    const float* mv_b     = (const float*)d_in[4];
    const float* mv_alpha = (const float*)d_in[5];
    const float* gate_w   = (const float*)d_in[6];
    const float* gate_b   = (const float*)d_in[7];
    const float* e_w1     = (const float*)d_in[8];
    const float* e_b1     = (const float*)d_in[9];
    const float* e_w2     = (const float*)d_in[10];
    const float* e_b2     = (const float*)d_in[11];
    const float* ln1_g    = (const float*)d_in[12];
    const float* ln1_b    = (const float*)d_in[13];
    const float* ln2_g    = (const float*)d_in[14];
    const float* ln2_b    = (const float*)d_in[15];
    const float* ln3_g    = (const float*)d_in[16];
    const float* ln3_b    = (const float*)d_in[17];
    float* out = (float*)d_out;

    char* ws = (char*)d_ws;
    size_t o = 0;
    auto alloc = [&](size_t bytes) -> void* {
        void* p = ws + o;
        o += (bytes + 255) & ~(size_t)255;
        return p;
    };
    // f32 buffers
    float* p1   = (float*)alloc((size_t)B_ * (S_ / 2) * D_ * 4);
    float* p2   = (float*)alloc((size_t)B_ * (S_ / 4) * D_ * 4);
    float* y0   = (float*)alloc((size_t)NTOK * D_ * 4);
    float* y1   = (float*)alloc((size_t)B_ * (S_ / 2) * D_ * 4);
    float* y2   = (float*)alloc((size_t)B_ * (S_ / 4) * D_ * 4);
    float* x1   = (float*)alloc((size_t)NTOK * D_ * 4);
    float* vbuf = (float*)alloc(3 * (size_t)NTOK * D_ * 4);
    float* x2   = (float*)alloc((size_t)NTOK * D_ * 4);
    float* ye   = (float*)alloc((size_t)CAPR * D_ * 4);
    // bf16 buffers
    __bf16* xb    = (__bf16*)alloc((size_t)NTOK * D_ * 2);
    __bf16* p1b   = (__bf16*)alloc((size_t)B_ * (S_ / 2) * D_ * 2);
    __bf16* p2b   = (__bf16*)alloc((size_t)B_ * (S_ / 4) * D_ * 2);
    __bf16* x1b   = (__bf16*)alloc((size_t)NTOK * D_ * 2);
    __bf16* x2b   = (__bf16*)alloc((size_t)NTOK * D_ * 2);
    __bf16* xg    = (__bf16*)alloc((size_t)CAPR * D_ * 2);
    __bf16* hbuf  = (__bf16*)alloc((size_t)CAPR * FF_ * 2);
    __bf16* mr_wt = (__bf16*)alloc(3 * (size_t)D_ * D_ * 2);
    __bf16* mv_wt = (__bf16*)alloc(3 * (size_t)D_ * D_ * 2);
    __bf16* w1t   = (__bf16*)alloc((size_t)E_ * D_ * FF_ * 2);
    __bf16* w2t   = (__bf16*)alloc((size_t)E_ * D_ * FF_ * 2);
    // routing state
    int*   cnt  = (int*)  alloc(E_ * 4);
    int*   off  = (int*)  alloc((E_ + 1) * 4);
    int*   texp = (int*)  alloc(NTOK * 2 * 4);
    int*   tslot= (int*)  alloc(NTOK * 2 * 4);
    float* tw   = (float*)alloc(NTOK * 2 * 4);
    (void)ws_size; (void)in_sizes; (void)n_in; (void)out_size;

    // 0) bf16 conversions: activations + transposed weights ([N][K] layout)
    k_cvt<<<(NTOK * D_ + 255) / 256, 256, 0, stream>>>(x, xb, NTOK * D_);
    { int n = 3 * D_ * D_;
      k_cvt_tr<<<(n + 255) / 256, 256, 0, stream>>>(mr_w, mr_wt, D_, D_, n);
      k_cvt_tr<<<(n + 255) / 256, 256, 0, stream>>>(mv_w, mv_wt, D_, D_, n); }
    { int n = E_ * D_ * FF_;
      k_cvt_tr<<<(n + 255) / 256, 256, 0, stream>>>(e_w1, w1t, D_, FF_, n);
      k_cvt_tr<<<(n + 255) / 256, 256, 0, stream>>>(e_w2, w2t, FF_, D_, n); }

    // 1) multiresolution pooling (+ bf16 copies)
    { int n1 = B_ * (S_ / 2) * D_;
      k_pool2<<<(n1 + 255) / 256, 256, 0, stream>>>(x, p1, n1, S_ / 2);
      int n2 = B_ * (S_ / 4) * D_;
      k_pool2<<<(n2 + 255) / 256, 256, 0, stream>>>(p1, p2, n2, S_ / 4);
      k_cvt<<<(n1 + 255) / 256, 256, 0, stream>>>(p1, p1b, n1);
      k_cvt<<<(n2 + 255) / 256, 256, 0, stream>>>(p2, p2b, n2); }

    // 2) three multires GEMMs (async-LDS + TDM bf16 WMMA)
    k_gemm<<<dim3(D_ / 128, NTOK / 128), 256, 0, stream>>>(
        xb,  D_, mr_wt + 0 * D_ * D_, D_, mr_b + 0 * D_, y0, (__bf16*)0, D_, NTOK, D_, 0);
    k_gemm<<<dim3(D_ / 128, (NTOK / 2) / 128), 256, 0, stream>>>(
        p1b, D_, mr_wt + 1 * D_ * D_, D_, mr_b + 1 * D_, y1, (__bf16*)0, D_, NTOK / 2, D_, 0);
    k_gemm<<<dim3(D_ / 128, (NTOK / 4) / 128), 256, 0, stream>>>(
        p2b, D_, mr_wt + 2 * D_ * D_, D_, mr_b + 2 * D_, y2, (__bf16*)0, D_, NTOK / 4, D_, 0);
    // 3) combine + LN1
    k_mrln1<<<NTOK / 8, 256, 0, stream>>>(x, y0, y1, y2, ln1_g, ln1_b, x1, x1b);
    // 4) three view GEMMs with fused GELU
    for (int v = 0; v < 3; ++v)
        k_gemm<<<dim3(D_ / 128, NTOK / 128), 256, 0, stream>>>(
            x1b, D_, mv_wt + (size_t)v * D_ * D_, D_, mv_b + v * D_,
            vbuf + (size_t)v * NTOK * D_, (__bf16*)0, D_, NTOK, D_, 1);
    // 5) softmax-weighted combine + LN2
    k_viewln2<<<NTOK / 8, 256, 0, stream>>>(x1, vbuf, mv_alpha, ln2_g, ln2_b, x2, x2b);
    // 6) top-2 MoE routing + padded segments + gather
    k_zero_f32<<<1, 32, 0, stream>>>((float*)cnt, E_);
    { int n = CAPR * D_ / 2;   // bf16 buffer zeroed as f32 words
      k_zero_f32<<<(n + 255) / 256, 256, 0, stream>>>((float*)xg, n); }
    k_route<<<NTOK / 8, 256, 0, stream>>>(x2, gate_w, gate_b, cnt, texp, tslot, tw);
    k_offsets<<<1, 1, 0, stream>>>(cnt, off);
    k_gather<<<NTOK * 2, 128, 0, stream>>>(x2b, texp, tslot, off, xg);
    // 7) expert FFN: xg @ w1 -> gelu -> h(bf16); h @ w2 -> ye(f32)
    k_moe_gemm<<<dim3(FF_ / 128, CAPR / 128), 256, 0, stream>>>(
        xg, D_, w1t, (size_t)D_ * FF_, D_, e_b1, FF_, (float*)0, hbuf, FF_, D_, 1, off);
    k_moe_gemm<<<dim3(D_ / 128, CAPR / 128), 256, 0, stream>>>(
        hbuf, FF_, w2t, (size_t)D_ * FF_, FF_, e_b2, D_, ye, (__bf16*)0, D_, FF_, 0, off);
    // 8) weighted combine + residual + LN3 -> out
    k_finln3<<<NTOK / 8, 256, 0, stream>>>(x2, ye, texp, tslot, off, tw,
                                           ln3_g, ln3_b, out);
}
